// SceneGraphEncoderLight_26671746908227
// MI455X (gfx1250) — matrix operand order repaired and checked by reading, not compiled, not run on hardware
//
#include <hip/hip_runtime.h>
#include <hip/hip_bf16.h>
#include <cstdint>

typedef _Float16 h16;
typedef __attribute__((ext_vector_type(16))) _Float16 v16h;
typedef __attribute__((ext_vector_type(8)))  _Float16 v8h;
typedef __attribute__((ext_vector_type(8)))  float    v8f;

#define N_NODES 50000
#define N_EDGES 800000
#define N_GRAPH 8
// order-preserving encoding of -inf for uint atomicMax-based float max
#define ENC_NEG_INF 0x007FFFFFu

// ---------------------------------------------------------------- helpers

__device__ __forceinline__ unsigned int encf(float f) {
  unsigned int b = __float_as_uint(f);
  return (b & 0x80000000u) ? ~b : (b | 0x80000000u);
}
__device__ __forceinline__ float decf(unsigned int u) {
  unsigned int b = (u & 0x80000000u) ? (u & 0x7fffffffu) : ~u;
  float f = __uint_as_float(b);
  if (!(fabsf(f) <= 3.0e38f)) f = 0.f;  // non-finite -> 0 (matches reference)
  return f;
}

// A fragment (16x32 f16, row-major source, lda in elements, lda % 8 == 0)
__device__ __forceinline__ v16h loadA(const h16* A, int lda, int r0, int kb, int lane) {
  const int m  = lane & 15;
  const int hi = (lane >> 4) & 1;
  const h16* p = A + (size_t)(r0 + m) * lda + kb * 32 + hi * 8;
  v8h lo = *(const v8h*)(p);
  v8h hh = *(const v8h*)(p + 16);
  v16h a;
#pragma unroll
  for (int i = 0; i < 8; ++i) { a[i] = lo[i]; a[i + 8] = hh[i]; }
  return a;
}

// B fragment from pre-packed weights (one contiguous 32B load per lane)
__device__ __forceinline__ v16h loadB(const h16* Bp, int nbCount, int kb, int nb, int lane) {
  const size_t t = (size_t)(kb * nbCount + nb);
  return *(const v16h*)(Bp + (t * 32 + lane) * 16);
}

#define WMMA_F16(A_, B_, C_) \
  __builtin_amdgcn_wmma_f32_16x16x32_f16(false, (A_), false, (B_), (short)0, (C_), false, false)

// ---------------------------------------------------------------- prep kernels

// pack weight W[K,N] (f32 row-major) into WMMA-B fragment order (f16)
__global__ void packB_kernel(const float* __restrict__ W, h16* __restrict__ Bp, int K, int N) {
  int idx = blockIdx.x * blockDim.x + threadIdx.x;
  if (idx >= K * N) return;
  int t      = idx >> 9;          // tile = 32 lanes * 16 elems
  int within = idx & 511;
  int lane   = within >> 4;
  int e      = within & 15;
  int nbc    = N >> 4;
  int kb = t / nbc, nb = t % nbc;
  int n = nb * 16 + (lane & 15);
  int k = kb * 32 + ((lane >> 4) & 1) * 8 + (e & 7) + (e >> 3) * 16;
  Bp[idx] = (h16)W[k * N + n];
}

__global__ void f32_to_f16_kernel(const float* __restrict__ in, h16* __restrict__ out, size_t n) {
  size_t i = (size_t)blockIdx.x * blockDim.x + threadIdx.x;
  size_t stride = (size_t)gridDim.x * blockDim.x;
  for (; i < n; i += stride) out[i] = (h16)in[i];
}

__global__ void fill_u32_kernel(unsigned int* __restrict__ p, unsigned int v, size_t n) {
  size_t i = (size_t)blockIdx.x * blockDim.x + threadIdx.x;
  size_t stride = (size_t)gridDim.x * blockDim.x;
  for (; i < n; i += stride) p[i] = v;
}

// ---------------------------------------------------------------- generic GEMM (wave = 16 rows)

template <int K, int NOUT>
__global__ void gemm_kernel(const h16* __restrict__ A, const h16* __restrict__ Bp,
                            const float* __restrict__ bias, float* __restrict__ outF,
                            h16* __restrict__ outH, int rows) {
  const int lane = threadIdx.x & 31;
  const int wave = threadIdx.x >> 5;
  const int tile = blockIdx.x * (blockDim.x >> 5) + wave;
  const int r0 = tile * 16;
  if (r0 >= rows) return;
  constexpr int NB = NOUT / 16;
  constexpr int KB = K / 32;
  v8f acc[NB];
#pragma unroll
  for (int i = 0; i < NB; ++i)
#pragma unroll
    for (int j = 0; j < 8; ++j) acc[i][j] = 0.f;
#pragma unroll
  for (int kb = 0; kb < KB; ++kb) {
    v16h a = loadA(A, K, r0, kb, lane);
#pragma unroll
    for (int nb = 0; nb < NB; ++nb) {
      v16h b = loadB(Bp, NB, kb, nb, lane);
      acc[nb] = WMMA_F16(a, b, acc[nb]);
    }
  }
  const int n = lane & 15;
  const int hi = (lane >> 4) & 1;
#pragma unroll
  for (int nb = 0; nb < NB; ++nb) {
    int col = nb * 16 + n;
    float bv = bias ? bias[col] : 0.f;
#pragma unroll
    for (int v = 0; v < 8; ++v) {
      size_t o = (size_t)(r0 + v + hi * 8) * NOUT + col;
      float val = acc[nb][v] + bv;
      outF[o] = val;
      if (outH) outH[o] = (h16)val;
    }
  }
}

// ---------------------------------------------------------------- fused 2-layer MLPs

// node encoder: relu(x@W1+b1)@W2+b2  (256 -> 128 -> 64)
__global__ void node_enc_kernel(const h16* __restrict__ xh, const h16* __restrict__ W1p,
                                const float* __restrict__ b1, const h16* __restrict__ W2p,
                                const float* __restrict__ b2, float* __restrict__ outF,
                                h16* __restrict__ outH, int rows) {
  __shared__ __align__(32) h16 lds[4][16 * 128];
  const int lane = threadIdx.x & 31;
  const int wave = threadIdx.x >> 5;
  const int r0 = (blockIdx.x * 4 + wave) * 16;
  const bool active = (r0 < rows);
  const int n = lane & 15, hi = (lane >> 4) & 1;
  v8f acc[8];
#pragma unroll
  for (int i = 0; i < 8; ++i)
#pragma unroll
    for (int j = 0; j < 8; ++j) acc[i][j] = 0.f;
  if (active) {
#pragma unroll
    for (int kb = 0; kb < 8; ++kb) {
      v16h a = loadA(xh, 256, r0, kb, lane);
#pragma unroll
      for (int nb = 0; nb < 8; ++nb) acc[nb] = WMMA_F16(a, loadB(W1p, 8, kb, nb, lane), acc[nb]);
    }
#pragma unroll
    for (int nb = 0; nb < 8; ++nb) {
      float bv = b1[nb * 16 + n];
#pragma unroll
      for (int v = 0; v < 8; ++v) {
        float t = acc[nb][v] + bv;
        lds[wave][(v + hi * 8) * 128 + nb * 16 + n] = (h16)(t > 0.f ? t : 0.f);
      }
    }
  }
  __syncthreads();
  if (active) {
    v8f acc2[4];
#pragma unroll
    for (int i = 0; i < 4; ++i)
#pragma unroll
      for (int j = 0; j < 8; ++j) acc2[i][j] = 0.f;
#pragma unroll
    for (int kb = 0; kb < 4; ++kb) {
      v16h a = loadA(&lds[wave][0], 128, 0, kb, lane);
#pragma unroll
      for (int nb = 0; nb < 4; ++nb) acc2[nb] = WMMA_F16(a, loadB(W2p, 4, kb, nb, lane), acc2[nb]);
    }
#pragma unroll
    for (int nb = 0; nb < 4; ++nb) {
      int col = nb * 16 + n;
      float bv = b2[col];
#pragma unroll
      for (int v = 0; v < 8; ++v) {
        size_t o = (size_t)(r0 + v + hi * 8) * 64 + col;
        float val = acc2[nb][v] + bv;
        outF[o] = val;
        outH[o] = (h16)val;
      }
    }
  }
}

// edge encoder: relu(ea@W1+b1)@W2+b2  (64 -> 64 -> 64)
__global__ void edge_enc_kernel(const h16* __restrict__ eaH, const h16* __restrict__ W1p,
                                const float* __restrict__ b1, const h16* __restrict__ W2p,
                                const float* __restrict__ b2, float* __restrict__ outF, int rows) {
  __shared__ __align__(32) h16 lds[4][16 * 64];
  const int lane = threadIdx.x & 31;
  const int wave = threadIdx.x >> 5;
  const int r0 = (blockIdx.x * 4 + wave) * 16;
  const bool active = (r0 < rows);
  const int n = lane & 15, hi = (lane >> 4) & 1;
  v8f acc[4];
#pragma unroll
  for (int i = 0; i < 4; ++i)
#pragma unroll
    for (int j = 0; j < 8; ++j) acc[i][j] = 0.f;
  if (active) {
#pragma unroll
    for (int kb = 0; kb < 2; ++kb) {
      v16h a = loadA(eaH, 64, r0, kb, lane);
#pragma unroll
      for (int nb = 0; nb < 4; ++nb) acc[nb] = WMMA_F16(a, loadB(W1p, 4, kb, nb, lane), acc[nb]);
    }
#pragma unroll
    for (int nb = 0; nb < 4; ++nb) {
      float bv = b1[nb * 16 + n];
#pragma unroll
      for (int v = 0; v < 8; ++v) {
        float t = acc[nb][v] + bv;
        lds[wave][(v + hi * 8) * 64 + nb * 16 + n] = (h16)(t > 0.f ? t : 0.f);
      }
    }
  }
  __syncthreads();
  if (active) {
    v8f acc2[4];
#pragma unroll
    for (int i = 0; i < 4; ++i)
#pragma unroll
      for (int j = 0; j < 8; ++j) acc2[i][j] = 0.f;
#pragma unroll
    for (int kb = 0; kb < 2; ++kb) {
      v16h a = loadA(&lds[wave][0], 64, 0, kb, lane);
#pragma unroll
      for (int nb = 0; nb < 4; ++nb) acc2[nb] = WMMA_F16(a, loadB(W2p, 4, kb, nb, lane), acc2[nb]);
    }
#pragma unroll
    for (int nb = 0; nb < 4; ++nb) {
      int col = nb * 16 + n;
      float bv = b2[col];
#pragma unroll
      for (int v = 0; v < 8; ++v)
        outF[(size_t)(r0 + v + hi * 8) * 64 + col] = acc2[nb][v] + bv;
    }
  }
}

// ---------------------------------------------------------------- GATv2 attention score
// wave = 16 edges; ee = edge_attr@We done with WMMA in-register (never materialized);
// s = leaky_relu(ee + xl[src] + xr[dst]); alpha = s . att[head]; atomicMax segment max.
template <int NOUT>  // 128 (4 heads) or 64 (2 heads)
__global__ void gat_alpha_kernel(const h16* __restrict__ eaH, const h16* __restrict__ Wep,
                                 const int* __restrict__ src, const int* __restrict__ dst,
                                 const float* __restrict__ xl, const float* __restrict__ xr,
                                 const float* __restrict__ att, float* __restrict__ alpha,
                                 unsigned int* __restrict__ amax) {
  constexpr int NB = NOUT / 16;
  constexpr int H = NOUT / 32;
  const int lane = threadIdx.x & 31;
  const int wave = threadIdx.x >> 5;
  const int e0 = (blockIdx.x * (blockDim.x >> 5) + wave) * 16;
  if (e0 >= N_EDGES) return;
  v8f acc[NB];
#pragma unroll
  for (int i = 0; i < NB; ++i)
#pragma unroll
    for (int j = 0; j < 8; ++j) acc[i][j] = 0.f;
#pragma unroll
  for (int kb = 0; kb < 2; ++kb) {
    v16h a = loadA(eaH, 64, e0, kb, lane);
#pragma unroll
    for (int nb = 0; nb < NB; ++nb) acc[nb] = WMMA_F16(a, loadB(Wep, NB, kb, nb, lane), acc[nb]);
  }
  const int n = lane & 15;
  const int hi = (lane >> 4) & 1;
  int myS[8], myD[8];
#pragma unroll
  for (int v = 0; v < 8; ++v) {
    int e = e0 + v + 8 * hi;
    myS[v] = src[e];
    myD[v] = dst[e];
  }
#pragma unroll
  for (int h = 0; h < H; ++h) {
    float aa[8];
#pragma unroll
    for (int v = 0; v < 8; ++v) aa[v] = 0.f;
#pragma unroll
    for (int nbi = 0; nbi < 2; ++nbi) {
      const int nb = h * 2 + nbi;
      const int col = nb * 16 + n;          // col = head*32 + d
      const float attv = att[col];
#pragma unroll
      for (int v = 0; v < 8; ++v) {
        float s = acc[nb][v] + xl[(size_t)myS[v] * NOUT + col] + xr[(size_t)myD[v] * NOUT + col];
        s = s > 0.f ? s : 0.2f * s;         // leaky relu
        float c = s * attv;
        c += __shfl_xor(c, 1, 32);
        c += __shfl_xor(c, 2, 32);
        c += __shfl_xor(c, 4, 32);
        c += __shfl_xor(c, 8, 32);          // sum over the 16 cols of this tile
        aa[v] += c;
      }
    }
    if (n == 0) {                            // lanes 0 and 16 own 8 edges each
#pragma unroll
      for (int v = 0; v < 8; ++v) {
        int e = e0 + v + 8 * hi;
        alpha[(size_t)e * H + h] = aa[v];
        atomicMax(&amax[(size_t)myD[v] * H + h], encf(aa[v]));
      }
    }
  }
}

// wave = 1 edge; ex = exp(alpha - amax[dst]); den += ex; agg[dst] += xl[src]*ex
template <int NOUT>
__global__ void gat_agg_kernel(const int* __restrict__ src, const int* __restrict__ dst,
                               const float* __restrict__ alpha, const unsigned int* __restrict__ amax,
                               const float* __restrict__ xl, float* __restrict__ den,
                               float* __restrict__ agg) {
  constexpr int H = NOUT / 32;
  const int lane = threadIdx.x & 31;
  const int e = blockIdx.x * (blockDim.x >> 5) + (threadIdx.x >> 5);
  if (e >= N_EDGES) return;
  const int s = src[e], d = dst[e];
#pragma unroll
  for (int h = 0; h < H; ++h) {
    float ex = __expf(alpha[(size_t)e * H + h] - decf(amax[(size_t)d * H + h]));
    if (lane == 0) atomicAdd(&den[(size_t)d * H + h], ex);
    int col = h * 32 + lane;
    atomicAdd(&agg[(size_t)d * NOUT + col], xl[(size_t)s * NOUT + col] * ex);
  }
}

// ---------------------------------------------------------------- node finalize (wave = node)

// GAT1: out = elu(LN(agg/den + bias))
__global__ void gat1_final_kernel(const float* __restrict__ agg, const float* __restrict__ den,
                                  const float* __restrict__ bias, const float* __restrict__ lng,
                                  const float* __restrict__ lnb, float* __restrict__ outF,
                                  h16* __restrict__ outH) {
  const int lane = threadIdx.x & 31;
  const int node = blockIdx.x * (blockDim.x >> 5) + (threadIdx.x >> 5);
  if (node >= N_NODES) return;
  float vals[4], s = 0.f;
#pragma unroll
  for (int i = 0; i < 4; ++i) {
    int col = i * 32 + lane;
    vals[i] = agg[(size_t)node * 128 + col] / (den[(size_t)node * 4 + i] + 1e-16f) + bias[col];
    s += vals[i];
  }
#pragma unroll
  for (int o = 1; o < 32; o <<= 1) s += __shfl_xor(s, o, 32);
  float mean = s * (1.f / 128.f), vs = 0.f;
#pragma unroll
  for (int i = 0; i < 4; ++i) { float d0 = vals[i] - mean; vs += d0 * d0; }
#pragma unroll
  for (int o = 1; o < 32; o <<= 1) vs += __shfl_xor(vs, o, 32);
  float inv = rsqrtf(vs * (1.f / 128.f) + 1e-5f);
#pragma unroll
  for (int i = 0; i < 4; ++i) {
    int col = i * 32 + lane;
    float y = (vals[i] - mean) * inv * lng[col] + lnb[col];
    y = y > 0.f ? y : (__expf(y) - 1.f);   // elu
    size_t o = (size_t)node * 128 + col;
    outF[o] = y;
    outH[o] = (h16)y;
  }
}

// GAT2: out = elu(LN(agg/den + bias) + skip)
__global__ void gat2_final_kernel(const float* __restrict__ agg, const float* __restrict__ den,
                                  const float* __restrict__ bias, const float* __restrict__ lng,
                                  const float* __restrict__ lnb, const float* __restrict__ xskip,
                                  float* __restrict__ outF, h16* __restrict__ outH) {
  const int lane = threadIdx.x & 31;
  const int node = blockIdx.x * (blockDim.x >> 5) + (threadIdx.x >> 5);
  if (node >= N_NODES) return;
  float vals[2], s = 0.f;
#pragma unroll
  for (int i = 0; i < 2; ++i) {
    int col = i * 32 + lane;
    vals[i] = agg[(size_t)node * 64 + col] / (den[(size_t)node * 2 + i] + 1e-16f) + bias[col];
    s += vals[i];
  }
#pragma unroll
  for (int o = 1; o < 32; o <<= 1) s += __shfl_xor(s, o, 32);
  float mean = s * (1.f / 64.f), vs = 0.f;
#pragma unroll
  for (int i = 0; i < 2; ++i) { float d0 = vals[i] - mean; vs += d0 * d0; }
#pragma unroll
  for (int o = 1; o < 32; o <<= 1) vs += __shfl_xor(vs, o, 32);
  float inv = rsqrtf(vs * (1.f / 64.f) + 1e-5f);
#pragma unroll
  for (int i = 0; i < 2; ++i) {
    int col = i * 32 + lane;
    size_t o = (size_t)node * 64 + col;
    float y = (vals[i] - mean) * inv * lng[col] + lnb[col] + xskip[o];
    y = y > 0.f ? y : (__expf(y) - 1.f);
    outF[o] = y;
    outH[o] = (h16)y;
  }
}

// ---------------------------------------------------------------- pooling

__global__ void gate_kernel(const float* __restrict__ ne, const float* __restrict__ w1,
                            const float* __restrict__ b1, const float* __restrict__ w2,
                            const float* __restrict__ b2, const int* __restrict__ batch,
                            float* __restrict__ gate, unsigned int* __restrict__ gmax) {
  int nidx = blockIdx.x * blockDim.x + threadIdx.x;
  if (nidx >= N_NODES) return;
  float hrow[64];
#pragma unroll
  for (int k = 0; k < 64; ++k) hrow[k] = ne[(size_t)nidx * 64 + k];
  float gsum = b2[0];
  for (int j = 0; j < 64; ++j) {
    float t = b1[j];
#pragma unroll
    for (int k = 0; k < 64; ++k) t += hrow[k] * w1[k * 64 + j];
    gsum += (t > 0.f ? t : 0.f) * w2[j];
  }
  gate[nidx] = gsum;
  atomicMax(&gmax[batch[nidx]], encf(gsum));
}

__global__ void pool_kernel(const float* __restrict__ ne, const float* __restrict__ gate,
                            const int* __restrict__ batch, const unsigned int* __restrict__ gmax,
                            float* __restrict__ gden, float* __restrict__ pooled) {
  const int lane = threadIdx.x & 31;
  const int node = blockIdx.x * (blockDim.x >> 5) + (threadIdx.x >> 5);
  if (node >= N_NODES) return;
  const int g = batch[node];
  float ex = __expf(gate[node] - decf(gmax[g]));
  if (lane == 0) atomicAdd(&gden[g], ex);
#pragma unroll
  for (int i = 0; i < 2; ++i) {
    int c = i * 32 + lane;
    atomicAdd(&pooled[g * 64 + c], ex * ne[(size_t)node * 64 + c]);
  }
}

__global__ void scene_kernel(const float* __restrict__ pooled, const float* __restrict__ gden,
                             const float* __restrict__ w1, const float* __restrict__ b1,
                             const float* __restrict__ w2, const float* __restrict__ b2,
                             float* __restrict__ z) {
  __shared__ float t1[N_GRAPH][64];
  int tid = threadIdx.x;
  for (int it = tid; it < N_GRAPH * 64; it += 256) {
    int g = it >> 6, j = it & 63;
    float dd = 1.f / (gden[g] + 1e-16f);
    float acc = b1[j];
    for (int k = 0; k < 64; ++k) acc += (pooled[g * 64 + k] * dd) * w1[k * 64 + j];
    t1[g][j] = acc > 0.f ? acc : 0.f;
  }
  __syncthreads();
  if (tid < N_GRAPH * 32) {
    int g = tid >> 5, j = tid & 31;
    float acc = b2[j];
    for (int k = 0; k < 64; ++k) acc += t1[g][k] * w2[k * 32 + j];
    z[g * 32 + j] = acc;
  }
}

// ---------------------------------------------------------------- launcher

extern "C" void kernel_launch(void* const* d_in, const int* in_sizes, int n_in,
                              void* d_out, int out_size, void* d_ws, size_t ws_size,
                              hipStream_t stream) {
  const float* x         = (const float*)d_in[0];
  const int*   eidx      = (const int*)d_in[1];
  const float* edge_attr = (const float*)d_in[2];
  const int*   batch     = (const int*)d_in[3];
  const float* ne_w1 = (const float*)d_in[4];  const float* ne_b1 = (const float*)d_in[5];
  const float* ne_w2 = (const float*)d_in[6];  const float* ne_b2 = (const float*)d_in[7];
  const float* ee_w1 = (const float*)d_in[8];  const float* ee_b1 = (const float*)d_in[9];
  const float* ee_w2 = (const float*)d_in[10]; const float* ee_b2 = (const float*)d_in[11];
  const float* g1_wl = (const float*)d_in[12]; const float* g1_bl = (const float*)d_in[13];
  const float* g1_wr = (const float*)d_in[14]; const float* g1_br = (const float*)d_in[15];
  const float* g1_we = (const float*)d_in[16]; const float* g1_att = (const float*)d_in[17];
  const float* g1_b  = (const float*)d_in[18]; const float* n1_g = (const float*)d_in[19];
  const float* n1_b  = (const float*)d_in[20];
  const float* g2_wl = (const float*)d_in[21]; const float* g2_bl = (const float*)d_in[22];
  const float* g2_wr = (const float*)d_in[23]; const float* g2_br = (const float*)d_in[24];
  const float* g2_we = (const float*)d_in[25]; const float* g2_att = (const float*)d_in[26];
  const float* g2_b  = (const float*)d_in[27]; const float* n2_g = (const float*)d_in[28];
  const float* n2_b  = (const float*)d_in[29];
  const float* rp_w  = (const float*)d_in[30]; const float* rp_b = (const float*)d_in[31];
  const float* gt_w1 = (const float*)d_in[32]; const float* gt_b1 = (const float*)d_in[33];
  const float* gt_w2 = (const float*)d_in[34]; const float* gt_b2 = (const float*)d_in[35];
  const float* se_w1 = (const float*)d_in[36]; const float* se_b1 = (const float*)d_in[37];
  const float* se_w2 = (const float*)d_in[38]; const float* se_b2 = (const float*)d_in[39];

  const int* src = eidx;
  const int* dst = eidx + N_EDGES;

  float* z_out  = (float*)d_out;                       // [8,32]
  float* ne_out = z_out + 256;                         // [N,64]
  float* er_out = ne_out + (size_t)N_NODES * 64;       // [E,64]

  // ---- workspace bump allocator (256B aligned)
  char* wptr = (char*)d_ws;
  auto alloc = [&](size_t bytes) -> void* {
    void* p = wptr;
    wptr += (bytes + 255) & ~(size_t)255;
    return p;
  };
  h16* xh    = (h16*)alloc((size_t)N_NODES * 256 * 2);
  h16* eaH   = (h16*)alloc((size_t)N_EDGES * 64 * 2);
  float* h0  = (float*)alloc((size_t)N_NODES * 64 * 4);
  h16* h0h   = (h16*)alloc((size_t)N_NODES * 64 * 2);
  float* xl1 = (float*)alloc((size_t)N_NODES * 128 * 4);
  float* xr1 = (float*)alloc((size_t)N_NODES * 128 * 4);
  float* al1 = (float*)alloc((size_t)N_EDGES * 4 * 4);
  unsigned int* am1 = (unsigned int*)alloc((size_t)N_NODES * 4 * 4);
  float* dn1 = (float*)alloc((size_t)N_NODES * 4 * 4);
  float* ag1 = (float*)alloc((size_t)N_NODES * 128 * 4);
  float* h1  = (float*)alloc((size_t)N_NODES * 128 * 4);
  h16* h1h   = (h16*)alloc((size_t)N_NODES * 128 * 2);
  float* xl2 = (float*)alloc((size_t)N_NODES * 64 * 4);
  float* xr2 = (float*)alloc((size_t)N_NODES * 64 * 4);
  float* xsk = (float*)alloc((size_t)N_NODES * 64 * 4);
  float* al2 = (float*)alloc((size_t)N_EDGES * 2 * 4);
  unsigned int* am2 = (unsigned int*)alloc((size_t)N_NODES * 2 * 4);
  float* dn2 = (float*)alloc((size_t)N_NODES * 2 * 4);
  float* ag2 = (float*)alloc((size_t)N_NODES * 64 * 4);
  h16* nhh   = (h16*)alloc((size_t)N_NODES * 64 * 2);
  float* gate = (float*)alloc((size_t)N_NODES * 4);
  unsigned int* gmax = (unsigned int*)alloc(N_GRAPH * 4);
  float* gden = (float*)alloc(N_GRAPH * 4);
  float* pooled = (float*)alloc(N_GRAPH * 64 * 4);
  h16* ne_w1p = (h16*)alloc(256 * 128 * 2);
  h16* ne_w2p = (h16*)alloc(128 * 64 * 2);
  h16* g1_wlp = (h16*)alloc(64 * 128 * 2);
  h16* g1_wrp = (h16*)alloc(64 * 128 * 2);
  h16* g1_wep = (h16*)alloc(64 * 128 * 2);
  h16* g2_wlp = (h16*)alloc(128 * 64 * 2);
  h16* g2_wrp = (h16*)alloc(128 * 64 * 2);
  h16* g2_wep = (h16*)alloc(64 * 64 * 2);
  h16* rp_wp  = (h16*)alloc(64 * 64 * 2);
  h16* ee_w1p = (h16*)alloc(64 * 64 * 2);
  h16* ee_w2p = (h16*)alloc(64 * 64 * 2);

  // ---- init
  hipMemsetAsync(dn1, 0, (size_t)N_NODES * 4 * 4, stream);
  hipMemsetAsync(ag1, 0, (size_t)N_NODES * 128 * 4, stream);
  hipMemsetAsync(dn2, 0, (size_t)N_NODES * 2 * 4, stream);
  hipMemsetAsync(ag2, 0, (size_t)N_NODES * 64 * 4, stream);
  hipMemsetAsync(gden, 0, N_GRAPH * 4, stream);
  hipMemsetAsync(pooled, 0, N_GRAPH * 64 * 4, stream);
  fill_u32_kernel<<<64, 256, 0, stream>>>(am1, ENC_NEG_INF, (size_t)N_NODES * 4);
  fill_u32_kernel<<<64, 256, 0, stream>>>(am2, ENC_NEG_INF, (size_t)N_NODES * 2);
  fill_u32_kernel<<<1, 32, 0, stream>>>(gmax, ENC_NEG_INF, N_GRAPH);

  // ---- pack weights + convert activations to f16
  auto pack = [&](const float* W, h16* Bp, int K, int N) {
    packB_kernel<<<(K * N + 255) / 256, 256, 0, stream>>>(W, Bp, K, N);
  };
  pack(ne_w1, ne_w1p, 256, 128); pack(ne_w2, ne_w2p, 128, 64);
  pack(g1_wl, g1_wlp, 64, 128);  pack(g1_wr, g1_wrp, 64, 128);  pack(g1_we, g1_wep, 64, 128);
  pack(g2_wl, g2_wlp, 128, 64);  pack(g2_wr, g2_wrp, 128, 64);  pack(g2_we, g2_wep, 64, 64);
  pack(rp_w, rp_wp, 64, 64);     pack(ee_w1, ee_w1p, 64, 64);   pack(ee_w2, ee_w2p, 64, 64);
  f32_to_f16_kernel<<<2048, 256, 0, stream>>>(x, xh, (size_t)N_NODES * 256);
  f32_to_f16_kernel<<<4096, 256, 0, stream>>>(edge_attr, eaH, (size_t)N_EDGES * 64);

  const int nodeTiles = N_NODES / 16;                 // 3125
  const int nodeBlks  = (nodeTiles + 3) / 4;
  const int edgeTiles = N_EDGES / 16;                 // 50000
  const int edgeBlks  = (edgeTiles + 3) / 4;
  const int edgeWaveBlks = (N_EDGES + 7) / 8;
  const int nodeWaveBlks = (N_NODES + 7) / 8;

  // ---- node encoder
  node_enc_kernel<<<nodeBlks, 128, 0, stream>>>(xh, ne_w1p, ne_b1, ne_w2p, ne_b2, h0, h0h, N_NODES);

  // ---- GAT layer 1 (4 heads x 32)
  gemm_kernel<64, 128><<<nodeBlks, 128, 0, stream>>>(h0h, g1_wlp, g1_bl, xl1, nullptr, N_NODES);
  gemm_kernel<64, 128><<<nodeBlks, 128, 0, stream>>>(h0h, g1_wrp, g1_br, xr1, nullptr, N_NODES);
  gat_alpha_kernel<128><<<edgeBlks, 128, 0, stream>>>(eaH, g1_wep, src, dst, xl1, xr1, g1_att, al1, am1);
  gat_agg_kernel<128><<<edgeWaveBlks, 256, 0, stream>>>(src, dst, al1, am1, xl1, dn1, ag1);
  gat1_final_kernel<<<nodeWaveBlks, 256, 0, stream>>>(ag1, dn1, g1_b, n1_g, n1_b, h1, h1h);

  // ---- GAT layer 2 (2 heads x 32) + residual
  gemm_kernel<128, 64><<<nodeBlks, 128, 0, stream>>>(h1h, g2_wlp, g2_bl, xl2, nullptr, N_NODES);
  gemm_kernel<128, 64><<<nodeBlks, 128, 0, stream>>>(h1h, g2_wrp, g2_br, xr2, nullptr, N_NODES);
  gemm_kernel<64, 64><<<nodeBlks, 128, 0, stream>>>(h0h, rp_wp, rp_b, xsk, nullptr, N_NODES);
  gat_alpha_kernel<64><<<edgeBlks, 128, 0, stream>>>(eaH, g2_wep, src, dst, xl2, xr2, g2_att, al2, am2);
  gat_agg_kernel<64><<<edgeWaveBlks, 256, 0, stream>>>(src, dst, al2, am2, xl2, dn2, ag2);
  gat2_final_kernel<<<nodeWaveBlks, 256, 0, stream>>>(ag2, dn2, g2_b, n2_g, n2_b, xsk, ne_out, nhh);

  // ---- edge encoder (writes edge_repr output directly)
  edge_enc_kernel<<<edgeBlks, 128, 0, stream>>>(eaH, ee_w1p, ee_b1, ee_w2p, ee_b2, er_out, N_EDGES);

  // ---- attentional pooling + scene encoder
  gate_kernel<<<(N_NODES + 255) / 256, 256, 0, stream>>>(ne_out, gt_w1, gt_b1, gt_w2, gt_b2, batch, gate, gmax);
  pool_kernel<<<nodeWaveBlks, 256, 0, stream>>>(ne_out, gate, batch, gmax, gden, pooled);
  scene_kernel<<<1, 256, 0, stream>>>(pooled, gden, se_w1, se_b1, se_w2, se_b2, z_out);
  (void)in_sizes; (void)n_in; (void)out_size; (void)ws_size;
}
